// PyramidVisionTransformer_14035953123630
// MI455X (gfx1250) — matrix operand order repaired and checked by reading, not compiled, not run on hardware
//
#include <hip/hip_runtime.h>
#include <hip/hip_bf16.h>

// ---------------------------------------------------------------------------
// CDNA5 (gfx1250) PVT forward. All GEMMs go through v_wmma_f32_16x16x32_f16,
// double-buffered LDS pipeline (1 barrier/k-step) + global_prefetch.
// ---------------------------------------------------------------------------

typedef __attribute__((ext_vector_type(16))) _Float16 v16h;
typedef __attribute__((ext_vector_type(8)))  _Float16 v8h;
typedef __attribute__((ext_vector_type(8)))  float    v8f;

#define EPI_BIAS 1
#define EPI_GELU 2
#define EPI_RES  4

__device__ __forceinline__ int imin(int a, int b) { return a < b ? a : b; }

// ---------------------------------------------------------------------------
// Batched/strided GEMM:  C[z] = alpha * A[z] @ B[z]  (+bias)(+GELU)(+C_old)
// A: [M,K] (lda), B: [K,N] (ldb) or [N,K] if transB, C: [M,N] (ldc). fp32 in
// memory, f16 WMMA compute. z -> (zb = z/nh, zh = z%nh) with independent
// batch/head strides so one launch covers B*heads attention batches.
//
// Block tile 64x64, 256 threads = 8 wave32s (2Mx4N); each wave owns a 32x16
// slab = two v_wmma_f32_16x16x32_f16 per 32-deep k-step. Software pipeline:
//   load tile k+1 (global_load_b128) -> compute tile k (ds_load_b128 + wmma)
//   -> store tile k+1 to alternate LDS buffer -> one barrier.
// ---------------------------------------------------------------------------
__global__ __launch_bounds__(256)
void gemm_wmma_f16(const float* __restrict__ A, int lda, long long sAb, long long sAh,
                   const float* __restrict__ Bm, int ldb, long long sBb, long long sBh,
                   int transB,
                   float* __restrict__ C, int ldc, long long sCb, long long sCh,
                   const float* __restrict__ bias,
                   int M, int N, int K, int nh, float alpha, int epi)
{
    __shared__ __align__(16) _Float16 As[2][64][32];   // [buf][m][k]
    __shared__ __align__(16) _Float16 Bt[2][64][32];   // [buf][n][k]

    const int zb = (int)blockIdx.z / nh;
    const int zh = (int)blockIdx.z % nh;
    const float* Ap = A  + (long long)zb * sAb + (long long)zh * sAh;
    const float* Bp = Bm + (long long)zb * sBb + (long long)zh * sBh;
    float*       Cp = C  + (long long)zb * sCb + (long long)zh * sCh;

    // uniform (per-block) vectorizability of the global loads
    const bool a16 = ((lda & 3) == 0) &&
                     ((((unsigned long long)(const void*)Ap) & 15ull) == 0ull);
    const bool b16v = ((ldb & 3) == 0) &&
                      ((((unsigned long long)(const void*)Bp) & 15ull) == 0ull);

    const int m0  = blockIdx.x * 64;
    const int n0  = blockIdx.y * 64;
    const int tid = threadIdx.x;
    const int wave = tid >> 5, lane = tid & 31;
    const int wm = wave >> 2, wn = wave & 3;     // 2 (M) x 4 (N) wave grid
    const int sec = lane >> 4, lm = lane & 15;   // half-wave sector, lane-in-16

    const bool fullM = (m0 + 64 <= M);
    const bool fullN = (n0 + 64 <= N);

    float fa[8], fb[8];                          // in-flight tile (registers)

    // ---- tile loaders: global -> registers (no LDS traffic yet) ----
    auto loadA = [&](int k0) {
        const int r  = tid >> 2;                 // 0..63
        const int c  = (tid & 3) * 8;            // 0/8/16/24
        const int gm = m0 + r;
        if (fullM & (k0 + 32 <= K)) {
            const float* src = Ap + (long long)gm * lda + (k0 + c);
            if (a16) {
                const float4 x0 = *(const float4*)(src);
                const float4 x1 = *(const float4*)(src + 4);
                fa[0] = x0.x; fa[1] = x0.y; fa[2] = x0.z; fa[3] = x0.w;
                fa[4] = x1.x; fa[5] = x1.y; fa[6] = x1.z; fa[7] = x1.w;
            } else {
                #pragma unroll
                for (int u = 0; u < 8; ++u) fa[u] = src[u];
            }
            __builtin_prefetch(src + 64, 0, 3);          // k+2 tile -> global_prefetch
        } else {
            const float* src = Ap + (long long)imin(gm, M - 1) * lda;
            #pragma unroll 1
            for (int u = 0; u < 8; ++u) {
                const int gk = k0 + c + u;
                float v = src[imin(gk, K - 1)];
                fa[u] = (gm < M && gk < K) ? v : 0.0f;
            }
        }
    };
    auto storeA = [&](int buf) {
        const int r = tid >> 2;
        const int c = (tid & 3) * 8;
        v8h h;
        #pragma unroll
        for (int u = 0; u < 8; ++u) h[u] = (_Float16)fa[u];
        *(v8h*)&As[buf][r][c] = h;
    };
    auto loadB = [&](int k0) {
        if (transB) {                            // B is [N,K]: contiguous along k
            const int n  = tid >> 2;
            const int c  = (tid & 3) * 8;
            const int gn = n0 + n;
            if (fullN & (k0 + 32 <= K)) {
                const float* src = Bp + (long long)gn * ldb + (k0 + c);
                if (b16v) {
                    const float4 x0 = *(const float4*)(src);
                    const float4 x1 = *(const float4*)(src + 4);
                    fb[0] = x0.x; fb[1] = x0.y; fb[2] = x0.z; fb[3] = x0.w;
                    fb[4] = x1.x; fb[5] = x1.y; fb[6] = x1.z; fb[7] = x1.w;
                } else {
                    #pragma unroll
                    for (int u = 0; u < 8; ++u) fb[u] = src[u];
                }
                __builtin_prefetch(src + 64, 0, 3);
            } else {
                const float* src = Bp + (long long)imin(gn, N - 1) * ldb;
                #pragma unroll 1
                for (int u = 0; u < 8; ++u) {
                    const int gk = k0 + c + u;
                    float v = src[imin(gk, K - 1)];
                    fb[u] = (gn < N && gk < K) ? v : 0.0f;
                }
            }
        } else {                                 // B is [K,N]: contiguous along n
            const int r  = tid >> 3;             // k row 0..31
            const int c  = (tid & 7) * 8;        // n offset 0..56
            const int gk = k0 + r;
            if (fullN & (k0 + 32 <= K)) {
                const float* src = Bp + (long long)gk * ldb + (n0 + c);
                if (b16v) {
                    const float4 x0 = *(const float4*)(src);
                    const float4 x1 = *(const float4*)(src + 4);
                    fb[0] = x0.x; fb[1] = x0.y; fb[2] = x0.z; fb[3] = x0.w;
                    fb[4] = x1.x; fb[5] = x1.y; fb[6] = x1.z; fb[7] = x1.w;
                } else {
                    #pragma unroll
                    for (int u = 0; u < 8; ++u) fb[u] = src[u];
                }
                __builtin_prefetch(src + (long long)32 * ldb, 0, 3);
            } else {
                const float* src = Bp + (long long)imin(gk, K - 1) * ldb;
                #pragma unroll 1
                for (int u = 0; u < 8; ++u) {
                    const int gn = n0 + c + u;
                    float v = src[imin(gn, N - 1)];
                    fb[u] = (gk < K && gn < N) ? v : 0.0f;
                }
            }
        }
    };
    auto storeB = [&](int buf) {
        if (transB) {
            const int n = tid >> 2;
            const int c = (tid & 3) * 8;
            v8h h;
            #pragma unroll
            for (int u = 0; u < 8; ++u) h[u] = (_Float16)fb[u];
            *(v8h*)&Bt[buf][n][c] = h;
        } else {
            const int r = tid >> 3;
            const int c = (tid & 7) * 8;
            #pragma unroll
            for (int u = 0; u < 8; ++u) Bt[buf][c + u][r] = (_Float16)fb[u];
        }
    };

    // ---- prologue: stage tile 0 ----
    loadA(0); loadB(0);
    storeA(0); storeB(0);
    __syncthreads();

    v8f acc0 = {}, acc1 = {};
    int cur = 0;

    for (int k0 = 0; k0 < K; k0 += 32) {
        const bool more = (k0 + 32) < K;        // uniform
        if (more) { loadA(k0 + 32); loadB(k0 + 32); }   // overlap with compute

        // ---- fragments (ISA 16-bit A 16x32 / B 32x16 lane layout, wave32)
        const v8h a0lo = *(const v8h*)&As[cur][wm * 32 + lm][sec * 8];
        const v8h a0hi = *(const v8h*)&As[cur][wm * 32 + lm][16 + sec * 8];
        const v8h a1lo = *(const v8h*)&As[cur][wm * 32 + 16 + lm][sec * 8];
        const v8h a1hi = *(const v8h*)&As[cur][wm * 32 + 16 + lm][16 + sec * 8];
        const v8h blo  = *(const v8h*)&Bt[cur][wn * 16 + lm][sec * 8];
        const v8h bhi  = *(const v8h*)&Bt[cur][wn * 16 + lm][16 + sec * 8];

        v16h af0, af1, bf;
        #pragma unroll
        for (int i = 0; i < 8; ++i) {
            af0[i] = a0lo[i]; af0[8 + i] = a0hi[i];
            af1[i] = a1lo[i]; af1[8 + i] = a1hi[i];
            bf[i]  = blo[i];  bf[8 + i]  = bhi[i];
        }

        acc0 = __builtin_amdgcn_wmma_f32_16x16x32_f16(
                   false, af0, false, bf, (short)0, acc0, false, false);
        acc1 = __builtin_amdgcn_wmma_f32_16x16x32_f16(
                   false, af1, false, bf, (short)0, acc1, false, false);

        if (more) {
            storeA(cur ^ 1); storeB(cur ^ 1);   // fill alternate buffer
            __syncthreads();                    // single barrier per k-step
        }
        cur ^= 1;
    }

    // ---- epilogue: VGPR r of tile t holds (m = wm*32 + t*16 + sec*8 + r, n = lm)
    const int gn = n0 + wn * 16 + lm;
    #pragma unroll
    for (int t = 0; t < 2; ++t) {
        const v8f& acc = t ? acc1 : acc0;
        #pragma unroll
        for (int r = 0; r < 8; ++r) {
            const int gm = m0 + wm * 32 + t * 16 + sec * 8 + r;
            if (gm < M && gn < N) {
                float v = acc[r] * alpha;
                if (epi & EPI_BIAS) v += bias[gn];
                if (epi & EPI_GELU) v = 0.5f * v * (1.0f + erff(v * 0.70710678118f));
                long long ci = (long long)gm * ldc + gn;
                if (epi & EPI_RES) v += Cp[ci];
                Cp[ci] = v;
            }
        }
    }
}

// ---------------------------------------------------------------------------
// Patch conv: kernel == stride (patch embed / spatial-reduction conv).
// layout 0: in is NCHW [B,Cin,H,W]; layout 1: in is token-major [B,H*W,Cin].
// out: token-major [B,(H/k)*(W/k),Cout] with bias.
// ---------------------------------------------------------------------------
__global__ __launch_bounds__(256)
void patch_conv(const float* __restrict__ in, const float* __restrict__ w,
                const float* __restrict__ bias, float* __restrict__ out,
                int Bn, int Cin, int H, int W, int Cout, int k, int layout)
{
    int Ho = H / k, Wo = W / k;
    long long total = (long long)Bn * Ho * Wo * Cout;
    long long idx = (long long)blockIdx.x * blockDim.x + threadIdx.x;
    if (idx >= total) return;
    int co = (int)(idx % Cout);
    long long t = idx / Cout;
    int sp = (int)(t % (Ho * Wo));
    int b  = (int)(t / (Ho * Wo));
    int ty = sp / Wo, tx = sp % Wo;

    float acc = bias[co];
    for (int c = 0; c < Cin; ++c)
        for (int ky = 0; ky < k; ++ky)
            for (int kx = 0; kx < k; ++kx) {
                int iy = ty * k + ky, ix = tx * k + kx;
                float xv = (layout == 0)
                    ? in[(((long long)b * Cin + c) * H + iy) * W + ix]
                    : in[((long long)b * H * W + (long long)iy * W + ix) * Cin + c];
                acc += xv * w[(((long long)co * Cin + c) * k + ky) * k + kx];
            }
    out[((long long)b * (Ho * Wo) + sp) * Cout + co] = acc;
}

// ---------------------------------------------------------------------------
// LayerNorm over last dim E; one block per row; in-place safe.
// ---------------------------------------------------------------------------
__global__ __launch_bounds__(256)
void layernorm_k(const float* __restrict__ in, const float* __restrict__ g,
                 const float* __restrict__ b, float* __restrict__ out, int E)
{
    __shared__ float s1[256], s2[256];
    long long row = blockIdx.x;
    const float* x = in + row * E;
    float* y = out + row * E;
    float sum = 0.f, sq = 0.f;
    for (int e = threadIdx.x; e < E; e += 256) { float v = x[e]; sum += v; sq += v * v; }
    s1[threadIdx.x] = sum; s2[threadIdx.x] = sq;
    __syncthreads();
    for (int s = 128; s > 0; s >>= 1) {
        if ((int)threadIdx.x < s) {
            s1[threadIdx.x] += s1[threadIdx.x + s];
            s2[threadIdx.x] += s2[threadIdx.x + s];
        }
        __syncthreads();
    }
    float mean = s1[0] / (float)E;
    float var  = s2[0] / (float)E - mean * mean;
    float inv  = rsqrtf(var + 1e-5f);
    for (int e = threadIdx.x; e < E; e += 256)
        y[e] = (x[e] - mean) * inv * g[e] + b[e];
}

// ---------------------------------------------------------------------------
// Row softmax (row length M <= 50) — one thread per row.
// ---------------------------------------------------------------------------
__global__ __launch_bounds__(256)
void softmax_rows(float* __restrict__ S, long long rows, int M)
{
    long long r = (long long)blockIdx.x * blockDim.x + threadIdx.x;
    if (r >= rows) return;
    float* p = S + r * M;
    float mx = -1e30f;
    for (int i = 0; i < M; ++i) mx = fmaxf(mx, p[i]);
    float s = 0.f;
    for (int i = 0; i < M; ++i) { float e = __expf(p[i] - mx); p[i] = e; s += e; }
    float inv = 1.0f / s;
    for (int i = 0; i < M; ++i) p[i] *= inv;
}

// x[b,t,e] += pos[t,e]
__global__ __launch_bounds__(256)
void add_pos_k(float* __restrict__ x, const float* __restrict__ pos,
               int Bn, int N, int E)
{
    long long total = (long long)Bn * N * E;
    long long i = (long long)blockIdx.x * blockDim.x + threadIdx.x;
    if (i >= total) return;
    long long ne = i % ((long long)N * E);
    x[i] += pos[ne];
}

// Stage-3: out[b,0,:] = cls + pos[0]; out[b,1+t,:] = tok[b,t,:] + pos[1+t]
__global__ __launch_bounds__(256)
void assemble_cls_k(const float* __restrict__ tok, const float* __restrict__ cls,
                    const float* __restrict__ pos, float* __restrict__ out,
                    int Bn, int Nt, int E)
{
    int Np = Nt + 1;
    long long total = (long long)Bn * Np * E;
    long long i = (long long)blockIdx.x * blockDim.x + threadIdx.x;
    if (i >= total) return;
    int e = (int)(i % E);
    long long t2 = i / E;
    int t = (int)(t2 % Np);
    int b = (int)(t2 / Np);
    float v = (t == 0) ? cls[e] : tok[((long long)b * Nt + (t - 1)) * E + e];
    out[i] = v + pos[(long long)t * E + e];
}

// ---------------------------------------------------------------------------
// Host side
// ---------------------------------------------------------------------------
static inline void launch_gemm(hipStream_t s,
    const float* A, int lda, long long sAb, long long sAh,
    const float* Bm, int ldb, long long sBb, long long sBh, int transB,
    float* C, int ldc, long long sCb, long long sCh, const float* bias,
    int M, int N, int K, int nh, int z, float alpha, int epi)
{
    dim3 g((M + 63) / 64, (N + 63) / 64, z);
    gemm_wmma_f16<<<g, dim3(256), 0, s>>>(A, lda, sAb, sAh, Bm, ldb, sBb, sBh,
                                          transB, C, ldc, sCb, sCh, bias,
                                          M, N, K, nh, alpha, epi);
}

struct BlkP {
    const float *n1_g, *n1_b, *wq, *wkv, *proj_w, *proj_b, *n2_g, *n2_b;
    const float *fc1_w, *fc1_b, *fc2_w, *fc2_b;
    const float *sr_w, *sr_b, *srn_g, *srn_b;
};
struct StageP {
    const float *pe_w, *pe_b, *pe_g, *pe_beta, *pos;
    BlkP blk[6];
};

extern "C" void kernel_launch(void* const* d_in, const int* in_sizes, int n_in,
                              void* d_out, int out_size, void* d_ws, size_t ws_size,
                              hipStream_t stream)
{
    (void)in_sizes; (void)out_size; (void)ws_size;
    const int EMB[4] = {64, 128, 256, 512};
    const int DEP[4] = {3, 4, 6, 3};
    const int NHh[4] = {1, 2, 4, 8};
    const int SRr[4] = {8, 4, 2, 1};
    const int Bn = 32;

    // ---- walk flattened parameter dict (insertion order) ----
    const float* P[512];
    int cnt = n_in < 512 ? n_in : 512;
    for (int i = 0; i < cnt; ++i) P[i] = (const float*)d_in[i];
    int pc = 0;
    const float* x_img = P[pc++];                 // [32,3,224,224]
    StageP SP[4];
    for (int i = 0; i < 4; ++i) {
        StageP& sp = SP[i];
        sp.pe_w = P[pc++]; sp.pe_b = P[pc++];
        sp.pe_g = P[pc++]; sp.pe_beta = P[pc++];
        sp.pos  = P[pc++];
        for (int j = 0; j < DEP[i]; ++j) {
            BlkP& bk = sp.blk[j];
            bk.n1_g = P[pc++]; bk.n1_b = P[pc++];
            bk.wq = P[pc++]; bk.wkv = P[pc++];
            bk.proj_w = P[pc++]; bk.proj_b = P[pc++];
            bk.n2_g = P[pc++]; bk.n2_b = P[pc++];
            bk.fc1_w = P[pc++]; bk.fc1_b = P[pc++];
            bk.fc2_w = P[pc++]; bk.fc2_b = P[pc++];
            if (SRr[i] > 1) {
                bk.sr_w = P[pc++]; bk.sr_b = P[pc++];
                bk.srn_g = P[pc++]; bk.srn_b = P[pc++];
            } else {
                bk.sr_w = bk.sr_b = bk.srn_g = bk.srn_b = nullptr;
            }
        }
    }
    const float* cls    = P[pc++];
    const float* norm_g = P[pc++];
    const float* norm_b = P[pc++];
    const float* head_w = P[pc++];
    const float* head_b = P[pc++];

    // ---- workspace bump allocation (~141 MB) ----
    const long long XMAX = 32LL * 3136 * 64;      // 6,422,528 floats
    float* w0 = (float*)d_ws;
    float* buf_x  = w0; w0 += XMAX;               // residual stream
    float* buf_y  = w0; w0 += XMAX;               // LN output (xn)
    float* buf_q  = w0; w0 += XMAX;               // queries
    float* buf_o  = w0; w0 += XMAX;               // attn out / conv staging (aliased)
    float* buf_xs = w0; w0 += (1LL << 20);        // SR-reduced tokens
    float* buf_kv = w0; w0 += (2LL << 20);        // [B,M,2E]
    float* buf_s  = w0; w0 += (5LL << 20);        // attn logits/probs [B,h,N,M]
    float* buf_h  = w0; w0 += (1LL << 20);        // MLP hidden (per-batch chunk)
    float* buf_z  = buf_o;                        // conv staging alias

    const float* cur = x_img;
    int curC = 3, curH = 224, layout = 0;

    for (int i = 0; i < 4; ++i) {
        const StageP& sp = SP[i];
        const int p  = (i == 0) ? 4 : 2;
        const int E  = EMB[i];
        const int Ho = curH / p;
        const int Nt = Ho * Ho;
        const int Np = Nt + ((i == 3) ? 1 : 0);
        const int nh = NHh[i], sr = SRr[i];
        const long long sNE = (long long)Np * E;

        // patch embed conv (+bias) -> token-major, then LN, (cls), +pos
        {
            long long tot = (long long)Bn * Nt * E;
            patch_conv<<<(unsigned)((tot + 255) / 256), 256, 0, stream>>>(
                cur, sp.pe_w, sp.pe_b, buf_z, Bn, curC, curH, curH, E, p, layout);
            if (i == 3) {
                layernorm_k<<<Bn * Nt, 256, 0, stream>>>(buf_z, sp.pe_g, sp.pe_beta, buf_z, E);
                long long tot2 = (long long)Bn * Np * E;
                assemble_cls_k<<<(unsigned)((tot2 + 255) / 256), 256, 0, stream>>>(
                    buf_z, cls, sp.pos, buf_x, Bn, Nt, E);
            } else {
                layernorm_k<<<Bn * Nt, 256, 0, stream>>>(buf_z, sp.pe_g, sp.pe_beta, buf_x, E);
                long long tot2 = (long long)Bn * Np * E;
                add_pos_k<<<(unsigned)((tot2 + 255) / 256), 256, 0, stream>>>(
                    buf_x, sp.pos, Bn, Np, E);
            }
        }

        for (int j = 0; j < DEP[i]; ++j) {
            const BlkP& bk = sp.blk[j];
            // xn = LN1(x)
            layernorm_k<<<Bn * Np, 256, 0, stream>>>(buf_x, bk.n1_g, bk.n1_b, buf_y, E);
            // q = xn @ wq                              [B,Np,E]
            launch_gemm(stream, buf_y, E, sNE, 0, bk.wq, E, 0, 0, 0,
                        buf_q, E, sNE, 0, nullptr, Np, E, E, 1, Bn, 1.0f, 0);
            // spatial reduction of keys/values
            const float* xs; long long sxs; int Mkv;
            if (sr > 1) {
                int Hs = Ho / sr; Mkv = Hs * Hs;
                long long ct = (long long)Bn * Mkv * E;
                patch_conv<<<(unsigned)((ct + 255) / 256), 256, 0, stream>>>(
                    buf_y, bk.sr_w, bk.sr_b, buf_xs, Bn, E, Ho, Ho, E, sr, 1);
                layernorm_k<<<Bn * Mkv, 256, 0, stream>>>(buf_xs, bk.srn_g, bk.srn_b, buf_xs, E);
                xs = buf_xs; sxs = (long long)Mkv * E;
            } else {
                xs = buf_y; Mkv = Np; sxs = sNE;
            }
            const long long sKV = (long long)Mkv * 2 * E;
            // kv = xs @ wkv                            [B,Mkv,2E]
            launch_gemm(stream, xs, E, sxs, 0, bk.wkv, 2 * E, 0, 0, 0,
                        buf_kv, 2 * E, sKV, 0, nullptr, Mkv, 2 * E, E, 1, Bn, 1.0f, 0);
            // S = (Q @ K^T) * d^-0.5    per (b,h); K head h = kv cols [h*64, h*64+64)
            const long long sSb = (long long)nh * Np * Mkv;
            const long long sSh = (long long)Np * Mkv;
            launch_gemm(stream, buf_q, E, sNE, 64, buf_kv, 2 * E, sKV, 64, 1,
                        buf_s, Mkv, sSb, sSh, nullptr, Np, Mkv, 64, nh, Bn * nh,
                        0.125f, 0);
            long long rows = (long long)Bn * nh * Np;
            softmax_rows<<<(unsigned)((rows + 255) / 256), 256, 0, stream>>>(buf_s, rows, Mkv);
            // O = P @ V    per (b,h); V head h = kv cols [(nh+h)*64, ...)
            launch_gemm(stream, buf_s, Mkv, sSb, sSh, buf_kv + (long long)nh * 64,
                        2 * E, sKV, 64, 0,
                        buf_o, E, sNE, 64, nullptr, Np, 64, Mkv, nh, Bn * nh, 1.0f, 0);
            // x = x + O @ proj_w + proj_b
            launch_gemm(stream, buf_o, E, sNE, 0, bk.proj_w, E, 0, 0, 0,
                        buf_x, E, sNE, 0, bk.proj_b, Np, E, E, 1, Bn, 1.0f,
                        EPI_BIAS | EPI_RES);
            // MLP: x = x + fc2(gelu(fc1(LN2(x))))   (hidden chunked per batch)
            layernorm_k<<<Bn * Np, 256, 0, stream>>>(buf_x, bk.n2_g, bk.n2_b, buf_y, E);
            for (int b = 0; b < Bn; ++b) {
                launch_gemm(stream, buf_y + (long long)b * sNE, E, 0, 0,
                            bk.fc1_w, 4 * E, 0, 0, 0,
                            buf_h, 4 * E, 0, 0, bk.fc1_b,
                            Np, 4 * E, E, 1, 1, 1.0f, EPI_BIAS | EPI_GELU);
                launch_gemm(stream, buf_h, 4 * E, 0, 0,
                            bk.fc2_w, E, 0, 0, 0,
                            buf_x + (long long)b * sNE, E, 0, 0, bk.fc2_b,
                            Np, E, 4 * E, 1, 1, 1.0f, EPI_BIAS | EPI_RES);
            }
        }
        cur = buf_x; curC = E; curH = Ho; layout = 1;
    }

    // final LN + classifier head on cls token: out[b,:] = LN(x)[b,0,:] @ head_w + head_b
    layernorm_k<<<Bn * 50, 256, 0, stream>>>(buf_x, norm_g, norm_b, buf_y, 512);
    launch_gemm(stream, buf_y, 50 * 512, 0, 0,        // row m = batch m's token 0
                head_w, 1000, 0, 0, 0,
                (float*)d_out, 1000, 0, 0, head_b,
                32, 1000, 512, 1, 1, 1.0f, EPI_BIAS);
}
